// RNN_78460462563720
// MI455X (gfx1250) — matrix operand order repaired and checked by reading, not compiled, not run on hardware
//
#include <hip/hip_runtime.h>
#include <hip/hip_bf16.h>

// ---------------------------------------------------------------------------
// GRU forward for MI455X (gfx1250, wave32, WMMA).
//
// B=256, T=1000, I=256, H=512.  16 persistent workgroups, one per 16-batch-row
// slice; each WG owns ALL 1536 gate columns so the 1000-step recurrence needs
// only workgroup barriers.  GEMMs: v_wmma_f32_16x16x32_f16 (f16 A/B, fp32
// accumulate); state + nonlinearities fp32.  Weights pre-packed to f16 in the
// exact WMMA B-fragment lane layout (L2-resident, 2.4 MB).
//
// Codegen battle log:
//  R1: loop-invariant B-frag loads hoisted & spilled -> launder base per iter.
//  R2: u64 launder lost addrspace -> flat_load (couples DScnt).
//  R3: AS(1)-typed loads, but address formed by 64-bit integer math ->
//      72 loop-invariant offset pairs hoisted & spilled (scratch_load_b64).
//  R4 (this): inttoptr the laundered SGPR base ONCE per iteration, then pure
//      AS(1) GEP indexing base[frag*32 + lane].  ISel then emits
//      global_load_b128 saddr(SGPR) + shared voffset(lane*32) + imm(frag*1024)
//      -- no 64-bit VALU adds, no spills, LOADcnt-only waits.
// ---------------------------------------------------------------------------

typedef __attribute__((ext_vector_type(16))) _Float16 v16h;
typedef __attribute__((ext_vector_type(8)))  float    v8f;

// Global-address-space view of a B-fragment (forces global_load selection).
typedef const __attribute__((address_space(1))) v16h gv16h;

#define B_ 256
#define T_ 1000
#define I_ 256
#define H_ 512
#define G_ 1536   // 3*H

// LDS partition sizes (floats / halves)
#define SM_FLOATS (16*1024 /*gates_rz*/ + 16*512 /*gates_in*/ + 16*512 /*gates_hn*/ \
                 + 16*512 /*hprev*/ + 1024 /*brz*/ + 512 /*bin*/ + 512 /*bhn*/)
#define SM_HALVES (16*I_ /*xfrag*/ + 16*H_ /*hfrag*/)
#define SMEM_BYTES (SM_FLOATS*4 + SM_HALVES*2)   // 196608 = 192 KB

// ---- B-matrix (K x 16, f16) WMMA layout: element (k, n) lives in
//      lane = n + 16*((k>>3)&1), half-slot j = ((k>>4)<<3) | (k&7)  (ISA 7.12.2, dense 32x16).
// ---- A-matrix (16 x 32, f16) layout: element (m, kk):
//      laneHi = (kk>>3)&1, lane = m + 16*laneHi, j = ((kk>>4)<<3) | (kk&7).

// Pre-pack W (G x K, fp32 row-major, gemm is act @ W^T) into f16 B-fragments:
// packed[((ntile*KS + ks)*32 + lane)*16 + j] = W[ntile*16 + (lane&15)]
//                                               [ks*32 + ((j>>3)<<4) + ((lane>>4)<<3) + (j&7)]
__global__ void gru_pack_w(const float* __restrict__ W, _Float16* __restrict__ o, int K) {
  const int KS = K >> 5;
  const int total = G_ * K;
  for (int idx = blockIdx.x * blockDim.x + threadIdx.x; idx < total;
       idx += gridDim.x * blockDim.x) {
    const int j    = idx & 15;
    const int lane = (idx >> 4) & 31;
    const int fi   = idx >> 9;          // ntile*KS + ks
    const int ks   = fi % KS;
    const int ntg  = fi / KS;
    const int row  = (ntg << 4) + (lane & 15);
    const int col  = (ks << 5) + ((j >> 3) << 4) + ((lane >> 4) << 3) + (j & 7);
    o[idx] = (_Float16)W[(size_t)row * K + col];
  }
}

// Pure AS(1) GEP: address = base + (frag*32 + lane) * 32B.
// frag part is a compile-time constant after unrolling -> 24-bit immediate;
// lane*32 is the single shared divergent voffset.
__device__ __forceinline__ v16h load_bfrag(gv16h* base, int frag, int lane) {
  return base[frag * 32 + lane];
}
__device__ __forceinline__ v16h load_afrag(const _Float16* frag, int ks, int lane) {
  return *reinterpret_cast<const v16h*>(frag + ks * 512 + lane * 16);
}
__device__ __forceinline__ v8f wmma_f16(v16h a, v16h b, v8f c) {
  return __builtin_amdgcn_wmma_f32_16x16x32_f16(false, a, false, b, (short)0, c,
                                                false, false);
}
__device__ __forceinline__ float sigm(float x) { return 1.0f / (1.0f + __expf(-x)); }

__global__ __launch_bounds__(1024, 1)
void gru_persistent(const float* __restrict__ x, const float* __restrict__ b_ih,
                    const float* __restrict__ b_hh, const float* __restrict__ h0,
                    const _Float16* __restrict__ wiP, const _Float16* __restrict__ whP,
                    float* __restrict__ out) {
  extern __shared__ char smem_raw[];
  float* gates_rz = reinterpret_cast<float*>(smem_raw);   // [16][1024] pre-act r|z
  float* gates_in = gates_rz + 16 * 1024;                 // [16][512]  i_n (input side)
  float* gates_hn = gates_in + 16 * 512;                  // [16][512]  h_n (recurrent side)
  float* hprev    = gates_hn + 16 * 512;                  // [16][512]  fp32 hidden state
  float* brz      = hprev + 16 * 512;                     // b_ih+b_hh for r,z (1024)
  float* bin_     = brz + 1024;                           // b_ih[n-gate] (512)
  float* bhn_     = bin_ + 512;                           // b_hh[n-gate] (512)
  _Float16* xfrag = reinterpret_cast<_Float16*>(bhn_ + 512); // x tile, A-frag layout
  _Float16* hfrag = xfrag + 16 * I_;                         // h tile, A-frag layout

  const int tid  = threadIdx.x;
  const int lane = tid & 31;
  const int wave = tid >> 5;
  const int b0   = blockIdx.x << 4;      // 16 batch rows per WG
  const int nt0  = wave * 3;             // 3 N-tiles (48 gate cols) per wave

  // ---- biases + initial state into LDS --------------------------------
  brz[tid & 1023] = b_ih[tid & 1023] + b_hh[tid & 1023];  // 1024 threads cover it
  if (tid < 512) { bin_[tid] = b_ih[1024 + tid]; bhn_[tid] = b_hh[1024 + tid]; }
  for (int i = tid; i < 16 * H_; i += 1024) {
    const int m = i >> 9, c = i & 511;
    const float h = h0[c];
    hprev[i] = h;
    const int kk = c & 31, ks = c >> 5;
    const int lhi = (kk >> 3) & 1, j = ((kk >> 4) << 3) | (kk & 7);
    hfrag[ks * 512 + (m + (lhi << 4)) * 16 + j] = (_Float16)h;
  }

  // Weight base addresses as u64; laundered once per t-iteration below.
  unsigned long long wiA = (unsigned long long)wiP;
  unsigned long long whA = (unsigned long long)whP;

  for (int t = 0; t < T_; ++t) {
    // ---- stage x[b0..b0+15, t, :] -> f16 A-fragments in LDS -----------
    {
      const int m  = tid >> 6;          // row 0..15
      const int k0 = (tid & 63) << 2;   // 4 consecutive k per thread
      const float4 v =
          *reinterpret_cast<const float4*>(x + ((size_t)(b0 + m) * T_ + t) * I_ + k0);
      const int kk = k0 & 31, ks = k0 >> 5;
      const int lhi = (kk >> 3) & 1, j0 = ((kk >> 4) << 3) | (kk & 7);
      _Float16* p = xfrag + ks * 512 + (m + (lhi << 4)) * 16 + j0;
      p[0] = (_Float16)v.x; p[1] = (_Float16)v.y;
      p[2] = (_Float16)v.z; p[3] = (_Float16)v.w;
      if (t + 1 < T_)  // warm L2/L0 for next timestep (global_prefetch_b8)
        __builtin_prefetch(x + ((size_t)(b0 + m) * T_ + (t + 1)) * I_ + k0, 0, 1);
    }
    __syncthreads();   // also guarantees previous step's elementwise is done

    // Launder weight bases (SGPR pair) so nothing derived from them can be
    // hoisted out of the t-loop, then convert to AS(1) pointers exactly once.
    asm volatile("" : "+s"(wiA), "+s"(whA));
    gv16h* wiB = reinterpret_cast<gv16h*>(wiA);
    gv16h* whB = reinterpret_cast<gv16h*>(whA);

    // ---- fused dual GEMM: acc_i = x_t @ W_ih^T ; acc_r = h @ W_hh^T ----
    v8f ai[3], ar[3];
#pragma unroll
    for (int q = 0; q < 3; ++q) { ai[q] = (v8f)(0.0f); ar[q] = (v8f)(0.0f); }
#pragma unroll
    for (int ks = 0; ks < I_ / 32; ++ks) {          // K = 256 input side
      const v16h a = load_afrag(xfrag, ks, lane);
#pragma unroll
      for (int q = 0; q < 3; ++q)
        ai[q] = wmma_f16(a, load_bfrag(wiB, (nt0 + q) * (I_ / 32) + ks, lane), ai[q]);
    }
#pragma unroll
    for (int ks = 0; ks < H_ / 32; ++ks) {          // K = 512 recurrent side
      const v16h a = load_afrag(hfrag, ks, lane);
#pragma unroll
      for (int q = 0; q < 3; ++q)
        ar[q] = wmma_f16(a, load_bfrag(whB, (nt0 + q) * (H_ / 32) + ks, lane), ar[q]);
    }

    // ---- scatter pre-activations to LDS (C/D layout: m = v + 8*(lane>=16),
    //      n = lane&15).  n-gate (tile >= 64) keeps input/recurrent separate.
    {
      const int mB = (lane >> 4) << 3;
      const int n  = lane & 15;
#pragma unroll
      for (int q = 0; q < 3; ++q) {
        const int ntg = nt0 + q;
        const int c   = (ntg << 4) + n;
        if (ntg < 64) {
#pragma unroll
          for (int v = 0; v < 8; ++v)
            gates_rz[(mB + v) * 1024 + c] = ai[q][v] + ar[q][v];
        } else {
          const int cn = c - 1024;
#pragma unroll
          for (int v = 0; v < 8; ++v) {
            gates_in[(mB + v) * 512 + cn] = ai[q][v];
            gates_hn[(mB + v) * 512 + cn] = ar[q][v];
          }
        }
      }
    }
    __syncthreads();

    // ---- elementwise GRU cell update (fp32), 8 elements / thread -------
#pragma unroll
    for (int i = 0; i < 8; ++i) {
      const int idx = tid + i * 1024;
      const int m = idx >> 9, c = idx & 511;
      const float r  = sigm(gates_rz[m * 1024 + c] + brz[c]);
      const float z  = sigm(gates_rz[m * 1024 + 512 + c] + brz[512 + c]);
      const float nn = tanhf(gates_in[m * 512 + c] + bin_[c] +
                             r * (gates_hn[m * 512 + c] + bhn_[c]));
      const float h  = hprev[m * 512 + c];
      const float hn = (1.0f - z) * nn + z * h;
      hprev[m * 512 + c] = hn;
      out[((size_t)(b0 + m) * T_ + t) * H_ + c] = hn;
      const int kk = c & 31, ks = c >> 5;
      const int lhi = (kk >> 3) & 1, j = ((kk >> 4) << 3) | (kk & 7);
      hfrag[ks * 512 + (m + (lhi << 4)) * 16 + j] = (_Float16)hn;
    }
  }

  __syncthreads();
  // ---- h_last tail: out[B*T*H ..] = final hidden state -----------------
  float* tail = out + (size_t)B_ * T_ * H_;
  for (int i = tid; i < 16 * H_; i += 1024) {
    const int m = i >> 9, c = i & 511;
    tail[(size_t)(b0 + m) * H_ + c] = hprev[m * 512 + c];
  }
}

extern "C" void kernel_launch(void* const* d_in, const int* in_sizes, int n_in,
                              void* d_out, int out_size, void* d_ws, size_t ws_size,
                              hipStream_t stream) {
  const float* x    = (const float*)d_in[0];
  const float* W_ih = (const float*)d_in[1];
  const float* W_hh = (const float*)d_in[2];
  const float* b_ih = (const float*)d_in[3];
  const float* b_hh = (const float*)d_in[4];
  const float* h0   = (const float*)d_in[5];
  float* out = (float*)d_out;

  // ws: packed f16 weights in WMMA B-fragment order (2.36 MB total)
  _Float16* wiP = (_Float16*)d_ws;          // G_*I_ halves
  _Float16* whP = wiP + (size_t)G_ * I_;    // G_*H_ halves

  gru_pack_w<<<512, 256, 0, stream>>>(W_ih, wiP, I_);
  gru_pack_w<<<512, 256, 0, stream>>>(W_hh, whP, H_);

  (void)hipFuncSetAttribute((const void*)gru_persistent,
                            hipFuncAttributeMaxDynamicSharedMemorySize, SMEM_BYTES);
  gru_persistent<<<B_ / 16, 1024, SMEM_BYTES, stream>>>(x, b_ih, b_hh, h0, wiP, whP,
                                                        out);
}
